// HierarchicalRNN_1322849927228
// MI455X (gfx1250) — matrix operand order repaired and checked by reading, not compiled, not run on hardware
//
#include <hip/hip_runtime.h>
#include <hip/hip_bf16.h>

// ---------------------------------------------------------------------------
// Types for CDNA5 WMMA (wave32): v_wmma_f32_16x16x32_bf16
// ---------------------------------------------------------------------------
typedef __attribute__((ext_vector_type(16))) __bf16 v16bf;
typedef __attribute__((ext_vector_type(8)))  float  v8f;

#define HSTR 264   // padded LDS row stride for h (528B = 132 dw = 4 mod 64 banks)

union Frag {
    unsigned u[8];
    uint4    q[2];
    v16bf    v;
};

__device__ __forceinline__ unsigned short f32_to_bf16(float f) {
    unsigned u = __builtin_bit_cast(unsigned, f);
    unsigned r = u + 0x7FFFu + ((u >> 16) & 1u);   // round-to-nearest-even
    return (unsigned short)(r >> 16);
}
__device__ __forceinline__ unsigned pk_bf16(float lo, float hi) {
    return ((unsigned)f32_to_bf16(hi) << 16) | (unsigned)f32_to_bf16(lo);
}
__device__ __forceinline__ float sigmoidf_(float v) {
    return 1.f / (1.f + __expf(-v));
}
__device__ __forceinline__ float tanhf_(float v) {
    float e = __expf(2.f * v);
    return 1.f - 2.f / (e + 1.f);    // safe at +/-inf
}
// Opaque per-iteration offset: blocks loop-invariant hoisting of the weight
// loads WITHOUT breaking kernarg provenance (loads stay global_load_b128).
__device__ __forceinline__ size_t launder_off() {
    size_t o = 0;
    asm volatile("" : "+s"(o));
    return o;
}

// load the 4 gate B-fragments of one K-chunk from global (packed layout)
__device__ __forceinline__ void loadg4(Frag f[4], const unsigned* base,
                                       int wave, int lane, int kc) {
#pragma unroll
    for (int gate = 0; gate < 4; ++gate) {
        const uint4* p = (const uint4*)(base +
            ((size_t)(gate * 16 + wave) * 8 + kc) * 256 + lane * 8);
        f[gate].q[0] = p[0];
        f[gate].q[1] = p[1];
    }
}
// load the 4 gate B-fragments of one K-chunk (kc<4) from staged LDS
__device__ __forceinline__ void loadl4(Frag f[4], const unsigned* wlds,
                                       int wave, int lane, int kc) {
#pragma unroll
    for (int gate = 0; gate < 4; ++gate) {
        const uint4* p = (const uint4*)(wlds +
            ((((wave * 4 + gate) * 4 + kc) * 32 + lane) * 8));
        f[gate].q[0] = p[0];
        f[gate].q[1] = p[1];
    }
}
__device__ __forceinline__ void wmma4(v8f acc[4], const Frag& a, const Frag b[4]) {
#pragma unroll
    for (int gate = 0; gate < 4; ++gate)
        acc[gate] = __builtin_amdgcn_wmma_f32_16x16x32_bf16(
            false, a.v, false, b[gate].v, (short)0, acc[gate], false, false);
}
// A-fragment (16x32 bf16) from a row-major bf16 row pointer
__device__ __forceinline__ Frag load_a(const unsigned short* hrow, int kc, int g) {
    Frag a;
    a.q[0] = *(const uint4*)(hrow + kc * 32 + g * 8);
    a.q[1] = *(const uint4*)(hrow + kc * 32 + 16 + g * 8);
    return a;
}

// ---------------------------------------------------------------------------
// Pack a row-major f32 weight matrix W[N x K] into the wave32 WMMA bf16
// B-fragment layout:
//   dst[((nt*KC + kc)*32 + lane)*8 + v];  n = nt*16 + (lane&15)
//   k = kc*32 + (v>>2)*16 + (lane>>4)*8 + 2(v&3) , +1
// ---------------------------------------------------------------------------
__global__ void pack_b_kernel(const float* __restrict__ W,
                              unsigned* __restrict__ dst,
                              int K, int total_dwords) {
    int idx = blockIdx.x * blockDim.x + threadIdx.x;
    if (idx >= total_dwords) return;
    int v    = idx & 7;
    int lane = (idx >> 3) & 31;
    int rest = idx >> 8;            // nt*KC + kc
    int KC   = K >> 5;
    int kc   = rest % KC;
    int nt   = rest / KC;
    int n    = nt * 16 + (lane & 15);
    int k0   = kc * 32 + ((v >> 2) << 4) + ((lane >> 4) << 3) + 2 * (v & 3);
    float lo = W[(size_t)n * K + k0];
    float hi = W[(size_t)n * K + k0 + 1];
    dst[idx] = pk_bf16(lo, hi);
}

// ---------------------------------------------------------------------------
// Layer 1: one block per sequence (one WGP), 16 waves, wave w owns hidden
// cols [16w,16w+16). h in padded LDS, c in registers. Whh kc0..3 staged in
// LDS (256KB); kc4..7 stream from L2 with explicit prefetch across the LDS
// compute phase.
// ---------------------------------------------------------------------------
__global__ __launch_bounds__(512) void lstm_layer1_kernel(
    const float* __restrict__ x,            // [32][256][16][32]
    const int*   __restrict__ flags,        // [32]
    const float* __restrict__ b_ih1,        // [2][1024]
    const float* __restrict__ b_hh1,        // [2][1024]
    const unsigned* __restrict__ wih_pack,  // 2 x [64][1][32][8]
    const unsigned* __restrict__ whh_pack,  // 2 x [64][8][32][8]
    unsigned short* __restrict__ h1_out)    // [32][16][256] bf16
{
    extern __shared__ char smem[];
    unsigned short (*hsh)[HSTR] = (unsigned short (*)[HSTR])smem;
    unsigned* wlds = (unsigned*)(smem + 16 * HSTR * 2);   // 256KB staged Whh

    const int seq  = blockIdx.x;
    const int tid  = threadIdx.x;
    const int wave = tid >> 5;
    const int lane = tid & 31;
    const int row  = lane & 15;     // batch row (A) / tile column n (C)
    const int g    = lane >> 4;     // K-half selector
    const int flag = flags[seq];

    // zero initial hidden state (16*HSTR shorts = 2112 dwords)
    for (int i = tid; i < 16 * HSTR / 2; i += 512)
        ((unsigned*)smem)[i] = 0u;

    const unsigned* wih = wih_pack + (size_t)flag * (64 * 1 * 256);
    const unsigned* whh = whh_pack + (size_t)flag * (64 * 8 * 256);

    // resident Wih fragments (K=32 -> 1 k-chunk) and scalar biases
    Frag  wf[4];
    float bias[4];
#pragma unroll
    for (int gate = 0; gate < 4; ++gate) {
        int nt = gate * 16 + wave;
        const uint4* p = (const uint4*)(wih + (size_t)nt * 256 + lane * 8);
        wf[gate].q[0] = p[0];
        wf[gate].q[1] = p[1];
        int col = flag * 1024 + gate * 256 + wave * 16 + row;
        bias[gate] = b_ih1[col] + b_hh1[col];
    }

    // stage this wave's Whh kc0..3 slice into LDS (one time)
#pragma unroll
    for (int gate = 0; gate < 4; ++gate) {
#pragma unroll
        for (int kc = 0; kc < 4; ++kc) {
            const uint4* src = (const uint4*)(whh +
                ((size_t)(gate * 16 + wave) * 8 + kc) * 256 + lane * 8);
            uint4 lo = src[0], hi = src[1];
            uint4* dst = (uint4*)(wlds + ((((wave * 4 + gate) * 4 + kc) * 32 + lane) * 8));
            dst[0] = lo;
            dst[1] = hi;
        }
    }

    float c[8], hreg[8];
#pragma unroll
    for (int r = 0; r < 8; ++r) { c[r] = 0.f; hreg[r] = 0.f; }

    __syncthreads();

    const float* xseq = x + (size_t)seq * (256 * 16 * 32);

#pragma unroll 1
    for (int t = 0; t < 256; ++t) {
        const unsigned* whh_t = whh + launder_off();

        // prefetch global K-chunks 4,5 (cover = entire LDS phase below)
        Frag p4[4], p5[4], p6[4], p7[4];
        loadg4(p4, whh_t, wave, lane, 4);
        loadg4(p5, whh_t, wave, lane, 5);

        // ---- A-fragment of x_t (convert f32 -> bf16 on the fly) ----
        const float* xr = xseq + (size_t)t * (16 * 32) + row * 32;
        Frag ax;
        {
            float4 a = *(const float4*)(xr + g * 8);
            float4 b = *(const float4*)(xr + g * 8 + 4);
            float4 e = *(const float4*)(xr + 16 + g * 8);
            float4 d = *(const float4*)(xr + 16 + g * 8 + 4);
            ax.u[0] = pk_bf16(a.x, a.y); ax.u[1] = pk_bf16(a.z, a.w);
            ax.u[2] = pk_bf16(b.x, b.y); ax.u[3] = pk_bf16(b.z, b.w);
            ax.u[4] = pk_bf16(e.x, e.y); ax.u[5] = pk_bf16(e.z, e.w);
            ax.u[6] = pk_bf16(d.x, d.y); ax.u[7] = pk_bf16(d.z, d.w);
        }

        // ---- gate chains start with inline-0 C matrix (bias added later) ----
        v8f zero = {};
        v8f acc[4];
#pragma unroll
        for (int gate = 0; gate < 4; ++gate)
            acc[gate] = __builtin_amdgcn_wmma_f32_16x16x32_bf16(
                false, ax.v, false, wf[gate].v, (short)0, zero, false, false);

        // ---- kc 0..3: weights from LDS (low latency) ----
#pragma unroll
        for (int kc = 0; kc < 4; ++kc) {
            Frag ah = load_a(&hsh[row][0], kc, g);
            Frag bw[4];
            loadl4(bw, wlds, wave, lane, kc);
            wmma4(acc, ah, bw);
        }

        // ---- kc 4..7: prefetched global weights ----
        loadg4(p6, whh_t, wave, lane, 6);
        { Frag ah = load_a(&hsh[row][0], 4, g); wmma4(acc, ah, p4); }
        loadg4(p7, whh_t, wave, lane, 7);
        { Frag ah = load_a(&hsh[row][0], 5, g); wmma4(acc, ah, p5); }
        { Frag ah = load_a(&hsh[row][0], 6, g); wmma4(acc, ah, p6); }
        { Frag ah = load_a(&hsh[row][0], 7, g); wmma4(acc, ah, p7); }

        // ---- elementwise LSTM cell update (bias folded in here) ----
#pragma unroll
        for (int r = 0; r < 8; ++r) {
            float iv = sigmoidf_(acc[0][r] + bias[0]);
            float fv = sigmoidf_(acc[1][r] + bias[1]);
            float gv = tanhf_(acc[2][r] + bias[2]);
            float ov = sigmoidf_(acc[3][r] + bias[3]);
            c[r]    = fv * c[r] + iv * gv;
            hreg[r] = ov * tanhf_(c[r]);
        }

        __syncthreads();   // all waves done reading old h
#pragma unroll
        for (int r = 0; r < 8; ++r)
            hsh[8 * g + r][wave * 16 + row] = f32_to_bf16(hreg[r]);
        __syncthreads();   // new h visible to all waves
    }

    // final hidden of this sequence -> layer-2 input buffer (bf16)
    unsigned short* ho = h1_out + (size_t)seq * (16 * 256);
#pragma unroll
    for (int r = 0; r < 8; ++r)
        ho[(size_t)(8 * g + r) * 256 + wave * 16 + row] = f32_to_bf16(hreg[r]);
}

// ---------------------------------------------------------------------------
// Layer 2: single block, 32 sequential steps over h1, then sigmoid head.
// Whh2 kc0..3 staged in LDS; Wih2 and Whh2 kc4..7 stream from L2.
// ---------------------------------------------------------------------------
__global__ __launch_bounds__(512) void lstm_layer2_kernel(
    const unsigned short* __restrict__ h1,   // [32][16][256] bf16
    const float* __restrict__ b_ih2,         // [1024]
    const float* __restrict__ b_hh2,         // [1024]
    const unsigned* __restrict__ wih2,       // [64][8][32][8]
    const unsigned* __restrict__ whh2,       // [64][8][32][8]
    const float* __restrict__ W_lin,         // [1][256]
    const float* __restrict__ b_lin,         // [1]
    float* __restrict__ out)                 // [16]
{
    extern __shared__ char smem[];
    unsigned short (*hsh)[HSTR] = (unsigned short (*)[HSTR])smem;
    float* h2f = (float*)(smem + 16 * HSTR * 2);                 // [16][260]
    unsigned* wlds = (unsigned*)(smem + 16 * HSTR * 2 + 16 * 260 * 4);

    const int tid  = threadIdx.x;
    const int wave = tid >> 5;
    const int lane = tid & 31;
    const int row  = lane & 15;
    const int g    = lane >> 4;

    for (int i = tid; i < 16 * HSTR / 2; i += 512)
        ((unsigned*)smem)[i] = 0u;

    float bias[4];
#pragma unroll
    for (int gate = 0; gate < 4; ++gate) {
        int col = gate * 256 + wave * 16 + row;
        bias[gate] = b_ih2[col] + b_hh2[col];
    }

    // stage Whh2 kc0..3 into LDS
#pragma unroll
    for (int gate = 0; gate < 4; ++gate) {
#pragma unroll
        for (int kc = 0; kc < 4; ++kc) {
            const uint4* src = (const uint4*)(whh2 +
                ((size_t)(gate * 16 + wave) * 8 + kc) * 256 + lane * 8);
            uint4 lo = src[0], hi = src[1];
            uint4* dst = (uint4*)(wlds + ((((wave * 4 + gate) * 4 + kc) * 32 + lane) * 8));
            dst[0] = lo;
            dst[1] = hi;
        }
    }

    float c[8], hreg[8];
#pragma unroll
    for (int r = 0; r < 8; ++r) { c[r] = 0.f; hreg[r] = 0.f; }

    __syncthreads();

#pragma unroll 1
    for (int t = 0; t < 32; ++t) {
        const unsigned* wih2_t = wih2 + launder_off();
        const unsigned* whh2_t = whh2 + launder_off();

        const unsigned short* xr = h1 + ((size_t)t * 16 + row) * 256;

        Frag p4[4], p5[4], p6[4], p7[4];
        loadg4(p4, whh2_t, wave, lane, 4);
        loadg4(p5, whh2_t, wave, lane, 5);

        v8f zero = {};
        v8f acc[4];

        // ---- kc 0..3: Wih2 from global (JIT), Whh2 from LDS ----
#pragma unroll
        for (int kc = 0; kc < 4; ++kc) {
            Frag bi[4];
            loadg4(bi, wih2_t, wave, lane, kc);
            Frag ax = load_a(xr, kc, g);
            Frag ah = load_a(&hsh[row][0], kc, g);
            Frag bh[4];
            loadl4(bh, wlds, wave, lane, kc);
            if (kc == 0) {
#pragma unroll
                for (int gate = 0; gate < 4; ++gate)
                    acc[gate] = __builtin_amdgcn_wmma_f32_16x16x32_bf16(
                        false, ax.v, false, bi[gate].v, (short)0, zero, false, false);
            } else {
                wmma4(acc, ax, bi);
            }
            wmma4(acc, ah, bh);
        }

        // ---- kc 4..7: prefetched Whh2 + JIT Wih2 ----
        loadg4(p6, whh2_t, wave, lane, 6);
#pragma unroll
        for (int i = 0; i < 2; ++i) {
            int kc = 4 + i;
            Frag bi[4];
            loadg4(bi, wih2_t, wave, lane, kc);
            Frag ax = load_a(xr, kc, g);
            Frag ah = load_a(&hsh[row][0], kc, g);
            wmma4(acc, ax, bi);
            wmma4(acc, ah, (i == 0) ? p4 : p5);
        }
        loadg4(p7, whh2_t, wave, lane, 7);
#pragma unroll
        for (int i = 0; i < 2; ++i) {
            int kc = 6 + i;
            Frag bi[4];
            loadg4(bi, wih2_t, wave, lane, kc);
            Frag ax = load_a(xr, kc, g);
            Frag ah = load_a(&hsh[row][0], kc, g);
            wmma4(acc, ax, bi);
            wmma4(acc, ah, (i == 0) ? p6 : p7);
        }

#pragma unroll
        for (int r = 0; r < 8; ++r) {
            float iv = sigmoidf_(acc[0][r] + bias[0]);
            float fv = sigmoidf_(acc[1][r] + bias[1]);
            float gv = tanhf_(acc[2][r] + bias[2]);
            float ov = sigmoidf_(acc[3][r] + bias[3]);
            c[r]    = fv * c[r] + iv * gv;
            hreg[r] = ov * tanhf_(c[r]);
        }

        __syncthreads();
#pragma unroll
        for (int r = 0; r < 8; ++r)
            hsh[8 * g + r][wave * 16 + row] = f32_to_bf16(hreg[r]);
        __syncthreads();
    }

    // final hidden in f32 for the output head (padded rows: stride 260)
#pragma unroll
    for (int r = 0; r < 8; ++r)
        h2f[(8 * g + r) * 260 + wave * 16 + row] = hreg[r];
    __syncthreads();

    if (tid < 16) {
        float a = b_lin[0];
        for (int k = 0; k < 256; ++k)
            a += h2f[tid * 260 + k] * W_lin[k];
        out[tid] = 1.f / (1.f + __expf(-a));
    }
}

// ---------------------------------------------------------------------------
// Host-side launch
// ---------------------------------------------------------------------------
extern "C" void kernel_launch(void* const* d_in, const int* in_sizes, int n_in,
                              void* d_out, int out_size, void* d_ws, size_t ws_size,
                              hipStream_t stream) {
    const float* x     = (const float*)d_in[0];
    const int*   flags = (const int*)  d_in[1];
    const float* W_ih1 = (const float*)d_in[2];
    const float* W_hh1 = (const float*)d_in[3];
    const float* b_ih1 = (const float*)d_in[4];
    const float* b_hh1 = (const float*)d_in[5];
    const float* W_ih2 = (const float*)d_in[6];
    const float* W_hh2 = (const float*)d_in[7];
    const float* b_ih2 = (const float*)d_in[8];
    const float* b_hh2 = (const float*)d_in[9];
    const float* W_lin = (const float*)d_in[10];
    const float* b_lin = (const float*)d_in[11];
    float* out = (float*)d_out;

    // workspace layout (bytes)
    char* ws = (char*)d_ws;
    const size_t WIH1_DW = 64 * 1 * 256;   // 16384 dwords per flavor
    const size_t WHH_DW  = 64 * 8 * 256;   // 131072 dwords per matrix
    unsigned* wih1p = (unsigned*)(ws);
    unsigned* whh1p = (unsigned*)(ws + 2 * WIH1_DW * 4);
    unsigned* wih2p = (unsigned*)(ws + 2 * WIH1_DW * 4 + 2 * WHH_DW * 4);
    unsigned* whh2p = (unsigned*)(ws + 2 * WIH1_DW * 4 + 3 * WHH_DW * 4);
    unsigned short* h1buf = (unsigned short*)(ws + 2 * WIH1_DW * 4 + 4 * WHH_DW * 4);

    const int TPB = 256;
    // pack layer-1 weights, both flag flavors
    pack_b_kernel<<<((int)WIH1_DW + TPB - 1) / TPB, TPB, 0, stream>>>(
        W_ih1,             wih1p,            32, (int)WIH1_DW);
    pack_b_kernel<<<((int)WIH1_DW + TPB - 1) / TPB, TPB, 0, stream>>>(
        W_ih1 + 1024 * 32, wih1p + WIH1_DW,  32, (int)WIH1_DW);
    pack_b_kernel<<<((int)WHH_DW + TPB - 1) / TPB, TPB, 0, stream>>>(
        W_hh1,              whh1p,           256, (int)WHH_DW);
    pack_b_kernel<<<((int)WHH_DW + TPB - 1) / TPB, TPB, 0, stream>>>(
        W_hh1 + 1024 * 256, whh1p + WHH_DW,  256, (int)WHH_DW);
    // pack layer-2 weights
    pack_b_kernel<<<((int)WHH_DW + TPB - 1) / TPB, TPB, 0, stream>>>(
        W_ih2, wih2p, 256, (int)WHH_DW);
    pack_b_kernel<<<((int)WHH_DW + TPB - 1) / TPB, TPB, 0, stream>>>(
        W_hh2, whh2p, 256, (int)WHH_DW);

    // dynamic LDS: layer1 = h(8448) + 256KB weights; layer2 adds h2f(16640)
    const size_t shmem1 = (size_t)16 * HSTR * 2 + 256 * 1024;
    const size_t shmem2 = (size_t)16 * HSTR * 2 + 16 * 260 * 4 + 256 * 1024;
    (void)hipFuncSetAttribute((const void*)lstm_layer1_kernel,
                              hipFuncAttributeMaxDynamicSharedMemorySize, (int)shmem1);
    (void)hipFuncSetAttribute((const void*)lstm_layer2_kernel,
                              hipFuncAttributeMaxDynamicSharedMemorySize, (int)shmem2);

    // layer 1: 32 independent sequences, one WGP each
    lstm_layer1_kernel<<<32, 512, shmem1, stream>>>(
        x, flags, b_ih1, b_hh1, wih1p, whh1p, h1buf);

    // layer 2 + output head: single block
    lstm_layer2_kernel<<<1, 512, shmem2, stream>>>(
        h1buf, b_ih2, b_hh2, wih2p, whh2p, W_lin, b_lin, out);
}
